// TextualNetBeta_17763984736877
// MI455X (gfx1250) — matrix operand order repaired and checked by reading, not compiled, not run on hardware
//
#include <hip/hip_runtime.h>
#include <cstdint>

// ---------------------------------------------------------------------------
// TextualNetBeta for MI455X (gfx1250, wave32).
// All GEMMs run on v_wmma_f32_16x16x32_f16. Weights are converted to f16 once;
// activation producers (LSTM gates, mask, LN, attention, GEMM epilogue) emit
// f16 mirrors so GEMMs consume f16 directly. Staging uses vectorized b64/b128
// copies plus global_prefetch of the next K tile; f32 accumulate/epilogue.
// ---------------------------------------------------------------------------
#define B_   32
#define LQ_  64
#define OC_  7
#define LO_  24
#define SP_  20
#define WS_  32
#define IN_  300
#define H_   512
#define FF_  2048
#define G4H_ 2048      // 4*H
#define BO_  (B_*OC_)  // 224
#define NEGf  (-1e9f)
#define NEGBIGf (-9e15f)

typedef __attribute__((ext_vector_type(16))) _Float16 v16h;
typedef __attribute__((ext_vector_type(8)))  float    v8f;

// f32 -> f16 elementwise convert
__global__ void cvt_f16(const float* __restrict__ src, _Float16* __restrict__ dst,
                        int n) {
  int i = blockIdx.x * blockDim.x + threadIdx.x;
  if (i < n) dst[i] = (_Float16)src[i];
}

// ---------------------------------------------------------------------------
// WMMA GEMM: C[M,N](ldc) = act( A[M,K] @ opB(B) + bias ), A/B in f16.
// Optional C16: f16 mirror of the output (same layout) for chained GEMMs.
// transB=0: B is [K,N] row-major.  transB=1: B is [N,K] row-major (W.T).
// Block = 256 threads (8 waves), block tile 64x64, wave tile 16x32
// (2 WMMAs sharing the A fragment), K tile 32.
// ---------------------------------------------------------------------------
#define KT 32
__global__ void __launch_bounds__(256)
gemm_wmma(const _Float16* __restrict__ A, const _Float16* __restrict__ Bm,
          const float* __restrict__ bias, float* __restrict__ C,
          _Float16* __restrict__ C16,
          int M, int N, int K, int transB, int act, int ldc)
{
  __shared__ _Float16 lA[64][KT];    // [m][k]
  __shared__ _Float16 lBt[64][KT];   // [n][k]  (B tile stored transposed)
  const int tid  = threadIdx.x;
  const int wave = tid >> 5;
  const int lane = tid & 31;
  const int bm = blockIdx.y * 64;
  const int bn = blockIdx.x * 64;
  const int wr = (wave >> 1) * 16;   // wave row offset in block tile
  const int wc = (wave & 1)  * 32;   // wave col offset in block tile

  v8f acc0 = {}, acc1 = {};
  for (int k0 = 0; k0 < K; k0 += KT) {
    const bool interior = (bm + 64 <= M) && (bn + 64 <= N) && (k0 + KT <= K);
    if (interior) {
      // ---- fast path: 8 halves per thread, b64 loads + b128 LDS store ----
      {
        int r = tid >> 2, c0 = (tid & 3) * 8;
        const _Float16* src = A + (size_t)(bm + r) * K + k0 + c0;
        uint2 u0 = *(const uint2*)(src);
        uint2 u1 = *(const uint2*)(src + 4);
        __builtin_prefetch(src + KT, 0, 0);      // next K tile of A
        uint4 u; u.x = u0.x; u.y = u0.y; u.z = u1.x; u.w = u1.y;
        *(uint4*)(&lA[r][c0]) = u;
      }
      if (transB) {
        int r = tid >> 2, c0 = (tid & 3) * 8;
        const _Float16* src = Bm + (size_t)(bn + r) * K + k0 + c0;
        uint2 u0 = *(const uint2*)(src);
        uint2 u1 = *(const uint2*)(src + 4);
        __builtin_prefetch(src + KT, 0, 0);      // next K tile of B
        uint4 u; u.x = u0.x; u.y = u0.y; u.z = u1.x; u.w = u1.y;
        *(uint4*)(&lBt[r][c0]) = u;
      } else {
        int kk = tid >> 3, n0 = (tid & 7) * 8;
        const _Float16* src = Bm + (size_t)(k0 + kk) * N + bn + n0;
        _Float16 tmp[8];
        *(uint2*)(tmp)     = *(const uint2*)(src);
        *(uint2*)(tmp + 4) = *(const uint2*)(src + 4);
        __builtin_prefetch(src + (size_t)KT * N, 0, 0);
#pragma unroll
        for (int i = 0; i < 8; ++i) lBt[n0 + i][kk] = tmp[i];
      }
    } else {
      // ---- edge fallback: elementwise with bounds checks ----
      for (int i = tid; i < 64 * KT; i += 256) {
        int r = i >> 5, c = i & 31;
        int gm = bm + r, gk = k0 + c;
        lA[r][c] = (gm < M && gk < K) ? A[(size_t)gm * K + gk] : (_Float16)0.f;
      }
      for (int i = tid; i < 64 * KT; i += 256) {
        int n = i >> 5, c = i & 31;
        int gn = bn + n, gk = k0 + c;
        _Float16 v = (_Float16)0.f;
        if (gn < N && gk < K)
          v = transB ? Bm[(size_t)gn * K + gk] : Bm[(size_t)gk * N + gn];
        lBt[n][c] = v;
      }
    }
    __syncthreads();

    // fragments per ISA 7.12.2 (16-bit A 16x32, B 32x16)
    const int m  = lane & 15;
    const int kh = lane >> 4;
    v16h af, bf0, bf1;
    const _Float16* ap  = &lA[wr + m][kh * 8];
#pragma unroll
    for (int j = 0; j < 8; ++j) { af[j] = ap[j]; af[j + 8] = ap[j + 16]; }
    const _Float16* bp0 = &lBt[wc + m][kh * 8];
    const _Float16* bp1 = &lBt[wc + 16 + m][kh * 8];
#pragma unroll
    for (int j = 0; j < 8; ++j) { bf0[j] = bp0[j]; bf0[j + 8] = bp0[j + 16]; bf1[j] = bp1[j]; bf1[j + 8] = bp1[j + 16]; }
    acc0 = __builtin_amdgcn_wmma_f32_16x16x32_f16(false, af, false, bf0,
                                                  (short)0, acc0, false, false);
    acc1 = __builtin_amdgcn_wmma_f32_16x16x32_f16(false, af, false, bf1,
                                                  (short)0, acc1, false, false);
    __syncthreads();
  }

  // store: lane holds n = lane%16; vgpr v -> m = v + 8*(lane/16)
#pragma unroll
  for (int sub = 0; sub < 2; ++sub) {
    const v8f* acc = sub ? &acc1 : &acc0;
    const int gn = bn + wc + sub * 16 + (lane & 15);
    if (gn < N) {
      float bv = bias ? bias[gn] : 0.f;
#pragma unroll
      for (int v = 0; v < 8; ++v) {
        int gm = bm + wr + v + (lane >> 4) * 8;
        if (gm < M) {
          float x = (*acc)[v] + bv;
          if (act == 1) x = fmaxf(x, 0.f);
          C[(size_t)gm * ldc + gn] = x;
          if (C16) C16[(size_t)gm * ldc + gn] = (_Float16)x;
        }
      }
    }
  }
}

// ---------------------------------------------------------------------------
// Small helper kernels
// ---------------------------------------------------------------------------
__global__ void vec_add2(const float* a, const float* b, float* o, int n) {
  int i = blockIdx.x * blockDim.x + threadIdx.x;
  if (i < n) o[i] = a[i] + b[i];
}

__global__ void fill_zero(float* p, size_t n) {
  size_t i = (size_t)blockIdx.x * blockDim.x + threadIdx.x;
  if (i < n) p[i] = 0.f;
}

__global__ void fill_zero_h(_Float16* p, size_t n) {
  size_t i = (size_t)blockIdx.x * blockDim.x + threadIdx.x;
  if (i < n) p[i] = (_Float16)0.f;
}

// mask[tok] = 1 if sum(|x_row|)==0
__global__ void compute_mask(const float* __restrict__ x, int* __restrict__ m,
                             int tok, int D) {
  int t = blockIdx.x * blockDim.x + threadIdx.x;
  if (t >= tok) return;
  const float* r = x + (size_t)t * D;
  float s = 0.f;
  for (int i = 0; i < D; ++i) s += fabsf(r[i]);
  m[t] = (s == 0.f) ? 1 : 0;
}

// apply pad-mask in place and emit f16 mirror
__global__ void mask_zero_dual(float* __restrict__ f, _Float16* __restrict__ fh,
                               const int* __restrict__ m, int tok) {
  size_t i = (size_t)blockIdx.x * blockDim.x + threadIdx.x;
  if (i >= (size_t)tok * H_) return;
  float v = m[i >> 9] ? 0.f : f[i];
  f[i] = v;
  fh[i] = (_Float16)v;
}

// LSTM pointwise: gates = pre[:,t,:] + G; update (h16,c); write h into hs[:,t,:]
__global__ void lstm_gates(const float* __restrict__ pre,
                           const float* __restrict__ G,
                           _Float16* __restrict__ h16, float* __restrict__ c,
                           float* __restrict__ hs, int Nseq, int T, int t) {
  int idx = blockIdx.x * blockDim.x + threadIdx.x;
  if (idx >= Nseq * H_) return;
  int n = idx >> 9, u = idx & 511;
  const float* pr = pre + ((size_t)(n * T + t)) * G4H_;
  const float* gr = G + (size_t)n * G4H_;
  float gi = pr[u]          + gr[u];
  float gf = pr[H_ + u]     + gr[H_ + u];
  float gg = pr[2 * H_ + u] + gr[2 * H_ + u];
  float go = pr[3 * H_ + u] + gr[3 * H_ + u];
  float si = 1.f / (1.f + expf(-gi));
  float sf = 1.f / (1.f + expf(-gf));
  float so = 1.f / (1.f + expf(-go));
  float cc = sf * c[idx] + si * tanhf(gg);
  c[idx] = cc;
  float hh = so * tanhf(cc);
  h16[idx] = (_Float16)hh;           // recurrent GEMM consumes f16 directly
  hs[((size_t)(n * T + t)) * H_ + u] = hh;
}

// masked softmax attention; one block per (qi, head, n); blockDim = 64 (=DH)
__global__ void __launch_bounds__(64)
attn_kernel(const float* __restrict__ q, const float* __restrict__ k,
            const float* __restrict__ v, const int* __restrict__ mask,
            float* __restrict__ o, _Float16* __restrict__ o16,
            int Nseq, int T) {
  const int qi = blockIdx.x, h = blockIdx.y, n = blockIdx.z;
  const int tid = threadIdx.x;
  __shared__ float pe[64], sv[64];
  const float* qp = q + ((size_t)(n * T + qi)) * H_ + h * 64;
  float x = -1e30f;
  if (tid < T) {
    const float* kp = k + ((size_t)(n * T + tid)) * H_ + h * 64;
    float d = 0.f;
    for (int e = 0; e < 64; ++e) d += qp[e] * kp[e];
    d *= 0.125f;                      // 1/sqrt(64)
    if (mask[n * T + tid]) d = NEGf;
    x = d;
  }
  sv[tid] = x; __syncthreads();
  for (int st = 32; st > 0; st >>= 1) {
    if (tid < st) sv[tid] = fmaxf(sv[tid], sv[tid + st]);
    __syncthreads();
  }
  float mx = sv[0]; __syncthreads();
  float e = (tid < T) ? expf(x - mx) : 0.f;
  pe[tid] = e; sv[tid] = e; __syncthreads();
  for (int st = 32; st > 0; st >>= 1) {
    if (tid < st) sv[tid] += sv[tid + st];
    __syncthreads();
  }
  float tot = sv[0];
  float acc = 0.f;
  for (int t = 0; t < T; ++t)
    acc += pe[t] * v[((size_t)(n * T + t)) * H_ + h * 64 + tid];
  size_t oi = ((size_t)(n * T + qi)) * H_ + h * 64 + tid;
  float r = acc / tot;
  o[oi] = r;
  o16[oi] = (_Float16)r;
}

// x = LN(x + p), emit f16 mirror; one block (128 thr) per token, 512 cols
__global__ void __launch_bounds__(128)
add_ln_kernel(float* __restrict__ x, const float* __restrict__ p,
              const float* __restrict__ g, const float* __restrict__ b,
              _Float16* __restrict__ x16) {
  const int t = blockIdx.x, tid = threadIdx.x;
  __shared__ float red[128];
  float* xr = x + (size_t)t * H_;
  _Float16* xh = x16 + (size_t)t * H_;
  const float* pr = p + (size_t)t * H_;
  float lv[4]; float s = 0.f;
#pragma unroll
  for (int i = 0; i < 4; ++i) { lv[i] = xr[tid + i * 128] + pr[tid + i * 128]; s += lv[i]; }
  red[tid] = s; __syncthreads();
  for (int st = 64; st > 0; st >>= 1) { if (tid < st) red[tid] += red[tid + st]; __syncthreads(); }
  float mean = red[0] / (float)H_; __syncthreads();
  s = 0.f;
#pragma unroll
  for (int i = 0; i < 4; ++i) { float d = lv[i] - mean; s += d * d; }
  red[tid] = s; __syncthreads();
  for (int st = 64; st > 0; st >>= 1) { if (tid < st) red[tid] += red[tid + st]; __syncthreads(); }
  float inv = rsqrtf(red[0] / (float)H_ + 1e-6f);
#pragma unroll
  for (int i = 0; i < 4; ++i) {
    int col = tid + i * 128;
    float y = g[col] * (lv[i] - mean) * inv + b[col];
    xr[col] = y;
    xh[col] = (_Float16)y;
  }
}

// dst[g*D+d] = sum_l src[(g*L+l)*D+d]
__global__ void sum_rows(const float* __restrict__ src, float* __restrict__ dst,
                         int Gn, int L, int D) {
  int idx = blockIdx.x * blockDim.x + threadIdx.x;
  if (idx >= Gn * D) return;
  int g = idx / D, d = idx - g * D;
  const float* s = src + (size_t)g * L * D + d;
  float acc = 0.f;
  for (int l = 0; l < L; ++l) acc += s[(size_t)l * D];
  dst[idx] = acc;
}

// out[(b*R+r)*S+s] = dot(a[(b*R+r)], c[(b*S+s)]) over D
__global__ void dots_kernel(const float* __restrict__ a,
                            const float* __restrict__ c,
                            float* __restrict__ out, int B, int R, int S, int D) {
  int idx = blockIdx.x * blockDim.x + threadIdx.x;
  if (idx >= B * R * S) return;
  int s = idx % S, br = idx / S, b = br / R;
  const float* ap = a + (size_t)br * D;
  const float* cp = c + (size_t)(b * S + s) * D;
  float acc = 0.f;
  for (int d = 0; d < D; ++d) acc += ap[d] * cp[d];
  out[idx] = acc;
}

__global__ void csf_len_kernel(const int* __restrict__ mc, float* __restrict__ len,
                               int BS, int W) {
  int idx = blockIdx.x * blockDim.x + threadIdx.x;
  if (idx >= BS) return;
  int cnt = 0;
  for (int w = 0; w < W; ++w) cnt += mc[idx * W + w] ? 0 : 1;
  len[idx] = cnt ? (float)cnt : NEGBIGf;
}

__global__ void argmax_div(const float* __restrict__ att, const float* __restrict__ len,
                           int* __restrict__ ix, int Rtot, int S, int rPerB) {
  int r = blockIdx.x * blockDim.x + threadIdx.x;
  if (r >= Rtot) return;
  int b = r / rPerB;
  float best = -1e30f; int bi = 0;
  for (int s = 0; s < S; ++s) {
    float v = att[r * S + s] / len[b * S + s];
    if (v > best) { best = v; bi = s; }
  }
  ix[r] = bi;
}

__global__ void opt_empty_kernel(const int* __restrict__ mo, int* __restrict__ oe,
                                 int BO, int L) {
  int idx = blockIdx.x * blockDim.x + threadIdx.x;
  if (idx >= BO) return;
  int e = 1;
  for (int l = 0; l < L; ++l) if (!mo[idx * L + l]) e = 0;
  oe[idx] = e;
}

// que_csf[b,o,w,d] = optEmpty ? 0 : fc[b, ixq[b], w, d]
__global__ void gather_q2c(const float* __restrict__ fc, const int* __restrict__ ixq,
                           const int* __restrict__ oe, float* __restrict__ dst) {
  size_t idx = (size_t)blockIdx.x * blockDim.x + threadIdx.x;
  if (idx >= (size_t)BO_ * WS_ * H_) return;
  int d = idx & 511, w = (idx >> 9) & 31, bo = idx >> 14, b = bo / OC_;
  dst[idx] = oe[bo] ? 0.f
           : fc[((size_t)(b * SP_ + ixq[b]) * WS_ + w) * H_ + d];
}

// opt_csf[b,o,w,d] = optEmpty ? 0 : fc[0, ixo[b,o], w, d]   (faithful csf[0])
__global__ void gather_o2c(const float* __restrict__ fc, const int* __restrict__ ixo,
                           const int* __restrict__ oe, float* __restrict__ dst) {
  size_t idx = (size_t)blockIdx.x * blockDim.x + threadIdx.x;
  if (idx >= (size_t)BO_ * WS_ * H_) return;
  int d = idx & 511, w = (idx >> 9) & 31, bo = idx >> 14;
  dst[idx] = oe[bo] ? 0.f
           : fc[((size_t)ixo[bo] * WS_ + w) * H_ + d];
}

__global__ void rowmask_feat(const float* __restrict__ x, int* __restrict__ m,
                             int rows) {
  int r = blockIdx.x * blockDim.x + threadIdx.x;
  if (r >= rows) return;
  const float* p = x + (size_t)r * H_;
  float s = 0.f;
  for (int d = 0; d < H_; ++d) s += fabsf(p[d]);
  m[r] = (s == 0.f) ? 1 : 0;
}

__global__ void bcast_feat4(const float* __restrict__ fq, float* __restrict__ dst) {
  size_t idx = (size_t)blockIdx.x * blockDim.x + threadIdx.x;
  if (idx >= (size_t)BO_ * LQ_ * H_) return;
  int d = idx & 511, l = (idx >> 9) & 63, bo = idx >> 15, b = bo / OC_;
  dst[idx] = fq[((size_t)(b * LQ_ + l)) * H_ + d];
}

__global__ void bcast_mask4(const int* __restrict__ mq, int* __restrict__ dst) {
  int idx = blockIdx.x * blockDim.x + threadIdx.x;
  if (idx >= BO_ * LQ_) return;
  int l = idx & 63, bo = idx >> 6, b = bo / OC_;
  dst[idx] = mq[b * LQ_ + l];
}

// out[t] = dot(A[t,:512], w) + b0
__global__ void colvec(const float* __restrict__ A, const float* __restrict__ w,
                       const float* __restrict__ b0, float* __restrict__ out, int tok) {
  int t = blockIdx.x * blockDim.x + threadIdx.x;
  if (t >= tok) return;
  const float* r = A + (size_t)t * H_;
  float acc = b0[0];
  for (int k = 0; k < H_; ++k) acc += r[k] * w[k];
  out[t] = acc;
}

// masked softmax over L per group (L <= 64), in place on a
__global__ void __launch_bounds__(64)
softmaxL(float* __restrict__ a, const int* __restrict__ mask, int L) {
  const int bo = blockIdx.x, tid = threadIdx.x;
  __shared__ float sv[64];
  float x = -1e30f;
  if (tid < L) { x = a[bo * L + tid]; if (mask[bo * L + tid]) x = NEGf; }
  sv[tid] = x; __syncthreads();
  for (int st = 32; st > 0; st >>= 1) { if (tid < st) sv[tid] = fmaxf(sv[tid], sv[tid + st]); __syncthreads(); }
  float mx = sv[0]; __syncthreads();
  float e = (tid < L) ? expf(x - mx) : 0.f;
  sv[tid] = e; __syncthreads();
  for (int st = 32; st > 0; st >>= 1) { if (tid < st) sv[tid] += sv[tid + st]; __syncthreads(); }
  if (tid < L) a[bo * L + tid] = e / sv[0];
}

// xa[bo*D+d] = sum_l w[bo*L+l] * x[(bo*L+l)*D+d]
__global__ void weighted_sum(const float* __restrict__ x, const float* __restrict__ w,
                             float* __restrict__ xa, int BO, int L) {
  int idx = blockIdx.x * blockDim.x + threadIdx.x;
  if (idx >= BO * H_) return;
  int bo = idx >> 9, d = idx & 511;
  float acc = 0.f;
  for (int l = 0; l < L; ++l)
    acc += w[bo * L + l] * x[((size_t)(bo * L + l)) * H_ + d];
  xa[idx] = acc;
}

__global__ void cls_kernel(const float* __restrict__ fus, const float* __restrict__ w,
                           const float* __restrict__ b0, const int* __restrict__ oe,
                           float* __restrict__ out, int BO) {
  int r = blockIdx.x * blockDim.x + threadIdx.x;
  if (r >= BO) return;
  const float* f = fus + (size_t)r * (4 * H_);
  float acc = b0[0];
  for (int k = 0; k < 4 * H_; ++k) acc += f[k] * w[k];
  out[r] = oe[r] ? NEGBIGf : acc;
}

// ---------------------------------------------------------------------------
// Host orchestration
// ---------------------------------------------------------------------------
extern "C" void kernel_launch(void* const* d_in, const int* in_sizes, int n_in,
                              void* d_out, int out_size, void* d_ws, size_t ws_size,
                              hipStream_t stream) {
  (void)in_sizes; (void)n_in; (void)out_size; (void)ws_size;

  const float* que = (const float*)d_in[0];   // [B,LQ,IN]
  const float* opt = (const float*)d_in[1];   // [B,OC,LO,IN]
  const float* csf = (const float*)d_in[2];   // [B,SP,WS,IN]
  // params in dict order
  const float* p_w_ih = (const float*)d_in[3];
  const float* p_w_hh = (const float*)d_in[4];
  const float* p_b_ih = (const float*)d_in[5];
  const float* p_b_hh = (const float*)d_in[6];
  const float* p_wq = (const float*)d_in[7];  const float* p_bq = (const float*)d_in[8];
  const float* p_wk = (const float*)d_in[9];  const float* p_bk = (const float*)d_in[10];
  const float* p_wv = (const float*)d_in[11]; const float* p_bv = (const float*)d_in[12];
  const float* p_wo = (const float*)d_in[13]; const float* p_bo = (const float*)d_in[14];
  const float* p_ln1g = (const float*)d_in[15]; const float* p_ln1b = (const float*)d_in[16];
  const float* p_fw1 = (const float*)d_in[17]; const float* p_fb1 = (const float*)d_in[18];
  const float* p_fw2 = (const float*)d_in[19]; const float* p_fb2 = (const float*)d_in[20];
  const float* p_ln2g = (const float*)d_in[21]; const float* p_ln2b = (const float*)d_in[22];
  const float* p_flw1 = (const float*)d_in[23]; const float* p_flb1 = (const float*)d_in[24];
  const float* p_flw2 = (const float*)d_in[25]; const float* p_flb2 = (const float*)d_in[26];
  const float* p_mrgw = (const float*)d_in[27]; const float* p_mrgb = (const float*)d_in[28];
  const float* p_clsw = (const float*)d_in[29]; const float* p_clsb = (const float*)d_in[30];

  float* out = (float*)d_out;                 // [B,OC] = 224

  // ---- workspace bump allocator (256B aligned) ----
  char* base = (char*)d_ws;
  size_t off = 0;
  auto allocf = [&](size_t nf) -> float* {
    float* p = (float*)(base + off); off += ((nf * 4 + 255) / 256) * 256; return p; };
  auto alloci = [&](size_t ni) -> int* {
    int* p = (int*)(base + off); off += ((ni * 4 + 255) / 256) * 256; return p; };
  auto alloch = [&](size_t nh) -> _Float16* {
    _Float16* p = (_Float16*)(base + off); off += ((nh * 2 + 255) / 256) * 256; return p; };

  const int TQ = B_ * LQ_;          // 2048 tokens
  const int TO = BO_ * LO_;         // 5376
  const int TC = B_ * SP_ * WS_;    // 20480
  const int TMAX = TC;

  float* biasL = allocf(G4H_);
  int* mq = alloci(TQ); int* mo = alloci(TO); int* mc = alloci(TC);
  float* fq = allocf((size_t)TQ * H_);
  float* fo = allocf((size_t)TO * H_);
  float* fc = allocf((size_t)TC * H_);
  float* pre = allocf((size_t)TMAX * G4H_);    // LSTM pre-acts; reused as FFN hidden
  float* G = allocf((size_t)(B_ * SP_) * G4H_);
  float* cbuf = allocf((size_t)(B_ * SP_) * H_);
  float* qb = allocf((size_t)TMAX * H_);
  float* kb = allocf((size_t)TMAX * H_);
  float* vb = allocf((size_t)TMAX * H_);
  float* ob = allocf((size_t)TMAX * H_);
  float* t1 = allocf((size_t)TMAX * H_);
  float* csum = allocf((size_t)B_ * SP_ * H_);
  float* qsum = allocf((size_t)B_ * H_);
  float* osum = allocf((size_t)BO_ * H_);
  float* attq = allocf(B_ * SP_);
  float* atto = allocf(BO_ * SP_);
  float* lenb = allocf(B_ * SP_);
  int* ixq = alloci(B_); int* ixo = alloci(BO_); int* optE = alloci(BO_);
  int* m4 = alloci(BO_ * LQ_);
  int* mqcsf = alloci(BO_ * WS_); int* mocsf = alloci(BO_ * WS_);
  float* aw = allocf(BO_ * LQ_);
  float* xa = allocf((size_t)BO_ * H_);
  float* fusion = allocf((size_t)BO_ * 4 * H_);
  // f16 weight copies (converted once; L2-resident thereafter)
  _Float16* h_wih = alloch((size_t)G4H_ * IN_);
  _Float16* h_whh = alloch((size_t)G4H_ * H_);
  _Float16* h_wq = alloch((size_t)H_ * H_);
  _Float16* h_wk = alloch((size_t)H_ * H_);
  _Float16* h_wv = alloch((size_t)H_ * H_);
  _Float16* h_wo = alloch((size_t)H_ * H_);
  _Float16* h_fw1 = alloch((size_t)H_ * FF_);
  _Float16* h_fw2 = alloch((size_t)FF_ * H_);
  _Float16* h_flw1 = alloch((size_t)H_ * H_);
  _Float16* h_mrg = alloch((size_t)H_ * H_);
  // f16 activation mirrors
  _Float16* h16  = alloch((size_t)(B_ * SP_) * H_);   // LSTM hidden state (f16)
  _Float16* fh   = alloch((size_t)TMAX * H_);         // f16 mirror of residual stream
  _Float16* oh   = alloch((size_t)TMAX * H_);         // f16 mirror of attention output
  _Float16* preh = alloch((size_t)TMAX * G4H_);       // f16 mirror of FFN hidden
  _Float16* ah   = alloch((size_t)TMAX * G4H_);       // generic cvt staging
  // aliases for the AttFlat stage (attention scratch is free by then)
  float* a1 = qb;       // [tok,512] hidden of AttFlat MLP
  float* que4 = kb;     // [B,OC,LQ,H]
  float* qcsf = vb;     // [B,OC,WS,H]
  float* ocsf = ob;     // [B,OC,WS,H]

  auto cvt = [&](const float* s, _Float16* d, int n) {
    cvt_f16<<<(n + 255) / 256, 256, 0, stream>>>(s, d, n);
  };
  // GEMM on f16 operands; optional f16 mirror of output
  auto gemmH = [&](const _Float16* Ah, const _Float16* Bh, const float* bias,
                   float* C, _Float16* C16,
                   int M, int N, int K, int transB, int act, int ldc) {
    dim3 g((N + 63) / 64, (M + 63) / 64);
    gemm_wmma<<<g, 256, 0, stream>>>(Ah, Bh, bias, C, C16, M, N, K, transB, act, ldc);
  };
  // GEMM with standalone f16 conversion of an f32 A operand
  auto gemmF = [&](const float* A32, const _Float16* Bh, const float* bias,
                   float* C, int M, int N, int K, int transB, int act, int ldc) {
    cvt(A32, ah, M * K);
    gemmH(ah, Bh, bias, C, nullptr, M, N, K, transB, act, ldc);
  };

  // convert weights once
  cvt(p_w_ih, h_wih, G4H_ * IN_);
  cvt(p_w_hh, h_whh, G4H_ * H_);
  cvt(p_wq, h_wq, H_ * H_);   cvt(p_wk, h_wk, H_ * H_);
  cvt(p_wv, h_wv, H_ * H_);   cvt(p_wo, h_wo, H_ * H_);
  cvt(p_fw1, h_fw1, H_ * FF_); cvt(p_fw2, h_fw2, FF_ * H_);
  cvt(p_flw1, h_flw1, H_ * H_); cvt(p_mrgw, h_mrg, H_ * H_);

  // combined LSTM bias (b_ih + b_hh)
  vec_add2<<<(G4H_ + 255) / 256, 256, 0, stream>>>(p_b_ih, p_b_hh, biasL, G4H_);

  // pad masks from raw embeddings
  compute_mask<<<(TQ + 127) / 128, 128, 0, stream>>>(que, mq, TQ, IN_);
  compute_mask<<<(TO + 127) / 128, 128, 0, stream>>>(opt, mo, TO, IN_);
  compute_mask<<<(TC + 127) / 128, 128, 0, stream>>>(csf, mc, TC, IN_);

  // ---- encoder: LSTM -> mask -> self-attention -> mask ----
  auto encode = [&](const float* X, const int* mask, float* f, int Nseq, int T) {
    const int tok = Nseq * T;
    gemmF(X, h_wih, biasL, pre, tok, G4H_, IN_, /*transB=*/1, 0, G4H_);
    fill_zero_h<<<(Nseq * H_ + 255) / 256, 256, 0, stream>>>(h16, (size_t)Nseq * H_);
    fill_zero<<<(Nseq * H_ + 255) / 256, 256, 0, stream>>>(cbuf, (size_t)Nseq * H_);
    for (int t = 0; t < T; ++t) {
      gemmH(h16, h_whh, nullptr, G, nullptr, Nseq, G4H_, H_, /*transB=*/1, 0, G4H_);
      lstm_gates<<<(Nseq * H_ + 255) / 256, 256, 0, stream>>>(pre, G, h16, cbuf,
                                                              f, Nseq, T, t);
    }
    mask_zero_dual<<<((size_t)tok * H_ + 255) / 256, 256, 0, stream>>>(f, fh, mask, tok);
    // self-attention (A operands consumed as f16 mirrors; no cvt passes)
    gemmH(fh, h_wq, p_bq, qb, nullptr, tok, H_, H_, 0, 0, H_);
    gemmH(fh, h_wk, p_bk, kb, nullptr, tok, H_, H_, 0, 0, H_);
    gemmH(fh, h_wv, p_bv, vb, nullptr, tok, H_, H_, 0, 0, H_);
    dim3 ag(T, 8, Nseq);
    attn_kernel<<<ag, 64, 0, stream>>>(qb, kb, vb, mask, ob, oh, Nseq, T);
    gemmH(oh, h_wo, p_bo, t1, nullptr, tok, H_, H_, 0, 0, H_);
    add_ln_kernel<<<tok, 128, 0, stream>>>(f, t1, p_ln1g, p_ln1b, fh);
    // FFN: hidden written in f32+f16 (preh feeds the second GEMM directly)
    gemmH(fh, h_fw1, p_fb1, pre, preh, tok, FF_, H_, 0, /*relu=*/1, FF_);
    gemmH(preh, h_fw2, p_fb2, t1, nullptr, tok, H_, FF_, 0, 0, H_);
    add_ln_kernel<<<tok, 128, 0, stream>>>(f, t1, p_ln2g, p_ln2b, fh);
    mask_zero_dual<<<((size_t)tok * H_ + 255) / 256, 256, 0, stream>>>(f, fh, mask, tok);
  };

  encode(que, mq, fq, B_, LQ_);
  encode(opt, mo, fo, BO_, LO_);
  encode(csf, mc, fc, B_ * SP_, WS_);

  // ---- find_context ----
  sum_rows<<<(B_ * SP_ * H_ + 255) / 256, 256, 0, stream>>>(fc, csum, B_ * SP_, WS_, H_);
  sum_rows<<<(B_ * H_ + 255) / 256, 256, 0, stream>>>(fq, qsum, B_, LQ_, H_);
  sum_rows<<<(BO_ * H_ + 255) / 256, 256, 0, stream>>>(fo, osum, BO_, LO_, H_);
  csf_len_kernel<<<(B_ * SP_ + 255) / 256, 256, 0, stream>>>(mc, lenb, B_ * SP_, WS_);
  dots_kernel<<<(B_ * SP_ + 255) / 256, 256, 0, stream>>>(qsum, csum, attq, B_, 1, SP_, H_);
  argmax_div<<<(B_ + 63) / 64, 64, 0, stream>>>(attq, lenb, ixq, B_, SP_, 1);
  dots_kernel<<<(BO_ * SP_ + 255) / 256, 256, 0, stream>>>(osum, csum, atto, B_, OC_, SP_, H_);
  argmax_div<<<(BO_ + 63) / 64, 64, 0, stream>>>(atto, lenb, ixo, BO_, SP_, OC_);
  opt_empty_kernel<<<(BO_ + 63) / 64, 64, 0, stream>>>(mo, optE, BO_, LO_);
  {
    size_t n = (size_t)BO_ * WS_ * H_;
    gather_q2c<<<(n + 255) / 256, 256, 0, stream>>>(fc, ixq, optE, qcsf);
    gather_o2c<<<(n + 255) / 256, 256, 0, stream>>>(fc, ixo, optE, ocsf);
  }
  rowmask_feat<<<(BO_ * WS_ + 127) / 128, 128, 0, stream>>>(qcsf, mqcsf, BO_ * WS_);
  rowmask_feat<<<(BO_ * WS_ + 127) / 128, 128, 0, stream>>>(ocsf, mocsf, BO_ * WS_);
  {
    size_t n = (size_t)BO_ * LQ_ * H_;
    bcast_feat4<<<(n + 255) / 256, 256, 0, stream>>>(fq, que4);
    bcast_mask4<<<(BO_ * LQ_ + 255) / 256, 256, 0, stream>>>(mq, m4);
  }

  // ---- AttFlat x4 -> fusion [BO, 2048] ----
  auto flat = [&](const float* x, const int* mask, int L, int col) {
    const int tok = BO_ * L;
    gemmF(x, h_flw1, p_flb1, a1, tok, H_, H_, 0, /*relu=*/1, H_);
    colvec<<<(tok + 127) / 128, 128, 0, stream>>>(a1, p_flw2, p_flb2, aw, tok);
    softmaxL<<<BO_, 64, 0, stream>>>(aw, mask, L);
    weighted_sum<<<(BO_ * H_ + 255) / 256, 256, 0, stream>>>(x, aw, xa, BO_, L);
    // merge GEMM writes its 512-wide slice straight into the fused feature
    cvt(xa, ah, BO_ * H_);
    gemmH(ah, h_mrg, p_mrgb, fusion + col, nullptr, BO_, H_, H_, 0, 0, 4 * H_);
  };
  flat(que4, m4,    LQ_, 0);
  flat(qcsf, mqcsf, WS_, H_);
  flat(fo,   mo,    LO_, 2 * H_);
  flat(ocsf, mocsf, WS_, 3 * H_);

  // ---- classifier ----
  cls_kernel<<<(BO_ + 63) / 64, 64, 0, stream>>>(fusion, p_clsw, p_clsb, optE, out, BO_);
}
